// HybridGraphQNN_65481071408267
// MI455X (gfx1250) — matrix operand order repaired and checked by reading, not compiled, not run on hardware
//
#include <hip/hip_runtime.h>
#include <hip/hip_bf16.h>
#include <math.h>

// ---------------------------------------------------------------------------
// HybridGraphQNN fused kernel for MI455X (gfx1250, wave32, WMMA).
//
// Phase 0 (prep):   normalize features, pack f16 WMMA fragments, build f16
//                   transposed feature table, zero agg buffer + zero page.
// Phase 1 (agg):    stream adjacency once (268 MB -> ~11.5 us roofline),
//                   non-temporal. Per 16x32 tile: Gram via
//                   v_wmma_f32_16x16x32_f16, square/threshold/select in-lane
//                   (Gram D layout == next WMMA A layout, no cross-lane ops),
//                   second WMMA accumulates adj_eff @ X in f32.
//                   Dead lanes read a zero page -> no exec masking in loop.
//                   Diagonal handled exactly by post-correction in phase 2.
// Phase 2 (mlp):    diag fix + tiny per-node MLP + sigmoid.
// ---------------------------------------------------------------------------

#define NN 8192
#define FF 8
#define JSLABS 16
#define ITILES (NN / 16)      // 512
#define THRESH 0.9f

typedef __attribute__((ext_vector_type(16))) _Float16     v16h;
typedef __attribute__((ext_vector_type(8)))  float        v8f;
typedef __attribute__((ext_vector_type(4)))  unsigned int u32x4;
typedef __attribute__((ext_vector_type(4)))  float        f32x4;

union V16H {                  // one WMMA 16-bit operand fragment (8 VGPRs)
  v16h      v;
  _Float16  h[16];
  u32x4     q[2];
};

union F8 {
  f32x4 q[2];
  float f[8];
};

// ---------------------------------------------------------------- phase 0 --
__global__ __launch_bounds__(256)
void qgnn_prep(const float* __restrict__ x,
               u32x4* __restrict__ packedXn,     // [NN] 8 f16 normalized feats
               _Float16* __restrict__ xT,        // [FF][NN] raw feats, f16
               float* __restrict__ agg,          // [NN][FF] zero-init
               float* __restrict__ zpage)        // 256B of zeros
{
  int i = blockIdx.x * blockDim.x + threadIdx.x;
  if (i >= NN) return;
  if (i < 64) zpage[i] = 0.f;
  float v[FF];
  float s = 0.f;
#pragma unroll
  for (int f = 0; f < FF; ++f) { v[f] = x[i * FF + f]; s += v[f] * v[f]; }
  float inv = 1.0f / (sqrtf(s) + 1e-12f);
  V16H pn;
  pn.q[0] = (u32x4)(0u); pn.q[1] = (u32x4)(0u);
#pragma unroll
  for (int f = 0; f < FF; ++f) {
    pn.h[f] = (_Float16)(v[f] * inv);       // normalized, for Gram/fidelity
    xT[f * NN + i] = (_Float16)v[f];        // raw, for adj_eff @ X
    agg[i * FF + f] = 0.f;
  }
  packedXn[i] = pn.q[0];
}

// ---------------------------------------------------------------- phase 1 --
__global__ __launch_bounds__(256)
void qgnn_aggregate(const float* __restrict__ adj,
                    const char* __restrict__ packed,   // u32x4 per node
                    const _Float16* __restrict__ xT,
                    const char* __restrict__ zp,       // 256B zero page
                    float* __restrict__ agg)
{
  const int lane = threadIdx.x & 31;
  const int wave = threadIdx.x >> 5;
  const int W    = blockIdx.x * 8 + wave;   // global wave id, < ITILES*JSLABS
  const int itile = W & (ITILES - 1);
  const int jslab = W / ITILES;
  const int i0   = itile * 16;
  const int col  = lane & 15;
  const int hi   = lane >> 4;               // 0 or 1 (lane half)
  const int irow = i0 + col;
  (void)irow;

  const bool lo  = lane < 16;
  const bool fok = col < FF;

  // Gram B fragment (i side), fixed: lane l<16 holds node i0+l's 8 normalized
  // f16 features in K=0..7; everything else zero (hi lanes read zero page).
  V16H bgram;
  bgram.q[1] = (u32x4)(0u);
  bgram.q[0] = *(const u32x4*)(lo ? packed + (size_t)(i0 + lane) * 16 : zp);

  // Gram A fragments: only q[0] is refreshed in the loop; q[1] stays zero.
  V16H af0, af1;
  af0.q[1] = (u32x4)(0u);
  af1.q[1] = (u32x4)(0u);

  const int jper = NN / JSLABS;             // 512
  const int jbeg = jslab * jper;

  const char* pa0 = lo ? packed + (size_t)(jbeg + lane) * 16      : zp;
  const char* pa1 = lo ? packed + (size_t)(jbeg + 16 + lane) * 16 : zp;
  const long  pstep = lo ? 32 * 16 : 0;

  const char* pb = fok ? (const char*)(xT + (size_t)col * NN + jbeg + 8 * hi)
                       : zp;
  const long  bstep = fok ? 64 : 0;

  const float* rp = adj + (size_t)irow * NN + jbeg + 8 * hi;

  v8f acc = {0.f, 0.f, 0.f, 0.f, 0.f, 0.f, 0.f, 0.f};
  const v8f czero = {0.f, 0.f, 0.f, 0.f, 0.f, 0.f, 0.f, 0.f};

  for (int it = 0; it < jper / 32; ++it) {
    // --- Gram A fragments for the two 16-wide j tiles (unconditional) ---
    af0.q[0] = *(const u32x4*)pa0;  pa0 += pstep;
    af1.q[0] = *(const u32x4*)pa1;  pa1 += pstep;

    // G'[j][i] = xn_j . xn_i ; D layout: lane gives i=i0+col, reg r gives
    // j = jb(+16) + 8*hi + r  -> matches next WMMA's A layout in-lane.
    v8f g0 = __builtin_amdgcn_wmma_f32_16x16x32_f16(
        false, af0.v, false, bgram.v, (short)0, czero, false, false);
    v8f g1 = __builtin_amdgcn_wmma_f32_16x16x32_f16(
        false, af1.v, false, bgram.v, (short)0, czero, false, false);

    // --- adjacency: streamed exactly once -> non-temporal ---
    F8 A0, A1;
    A0.q[0] = __builtin_nontemporal_load((const f32x4*)(rp + 0));
    A0.q[1] = __builtin_nontemporal_load((const f32x4*)(rp + 4));
    A1.q[0] = __builtin_nontemporal_load((const f32x4*)(rp + 16));
    A1.q[1] = __builtin_nontemporal_load((const f32x4*)(rp + 20));
    rp += 32;

    // --- fuse: adj_eff = (g^2 >= 0.9) ? 1.0 : adj  (== max(adj, fid_adj),
    //     since adj in [0,1)). Diagonal always hits; fixed in phase 2. ---
    V16H aout;
#pragma unroll
    for (int r = 0; r < 8; ++r) {
      float g  = g0[r];
      aout.h[r]     = (_Float16)((g * g >= THRESH) ? 1.0f : A0.f[r]);
      float g2 = g1[r];
      aout.h[8 + r] = (_Float16)((g2 * g2 >= THRESH) ? 1.0f : A1.f[r]);
    }

    // --- B fragment: transposed raw features, K=j (32), N=f (8 live cols) ---
    V16H bout;
    bout.q[0] = *(const u32x4*)(pb + 0);
    bout.q[1] = *(const u32x4*)(pb + 32);
    pb += bstep;

    // out_tile(16i x 16f) += adj_eff(16i x 32j) @ X(32j x 16f), f32 accum
    acc = __builtin_amdgcn_wmma_f32_16x16x32_f16(
        false, aout.v, false, bout.v, (short)0, acc, false, false);
  }

  // D layout: lane -> f = col, reg e -> i = i0 + e + 8*hi
  if (fok) {
#pragma unroll
    for (int e = 0; e < 8; ++e) {
      int io = i0 + e + 8 * hi;
      atomicAdd(&agg[io * FF + col], acc[e]);
    }
  }
}

// ---------------------------------------------------------------- phase 2 --
template <int IN, int OUT>
__device__ inline void lin_tanh(const float* h, const float* __restrict__ Wt,
                                const float* __restrict__ b, float* o) {
#pragma unroll
  for (int n = 0; n < OUT; ++n) {
    float t = b[n];
#pragma unroll
    for (int k = 0; k < IN; ++k) t += h[k] * Wt[k * OUT + n];
    o[n] = tanhf(t);
  }
}

__global__ __launch_bounds__(256)
void qgnn_mlp(const float* __restrict__ agg,
              const float* __restrict__ adj,
              const float* __restrict__ x,
              const float* Wfm, const float* bfm,
              const float* Wc1, const float* bc1,
              const float* Wp1, const float* bp1,
              const float* Wc2, const float* bc2,
              const float* Wp2, const float* bp2,
              const float* Wc3, const float* bc3,
              const float* Wh,  const float* bh,
              float* __restrict__ out)
{
  int i = blockIdx.x * blockDim.x + threadIdx.x;
  if (i >= NN) return;
  float h0[8], h1[16], h2[16], h3[12], h4[8], h5[4], h6[4];
  // exact diagonal correction: loop used eff(i,i)=1.0, true value is adj[i][i]
  float dcorr = adj[(size_t)i * NN + i] - 1.0f;
#pragma unroll
  for (int f = 0; f < 8; ++f) h0[f] = agg[i * 8 + f] + dcorr * x[i * 8 + f];
  lin_tanh<8, 16>(h0, Wfm, bfm, h1);
  lin_tanh<16, 16>(h1, Wc1, bc1, h2);
  lin_tanh<16, 12>(h2, Wp1, bp1, h3);
  lin_tanh<12, 8>(h3, Wc2, bc2, h4);
  lin_tanh<8, 4>(h4, Wp2, bp2, h5);
  lin_tanh<4, 4>(h5, Wc3, bc3, h6);
  float z = bh[0];
#pragma unroll
  for (int k = 0; k < 4; ++k) z += h6[k] * Wh[k];
  out[i] = 1.0f / (1.0f + expf(-z));
}

// ---------------------------------------------------------------- launch ---
extern "C" void kernel_launch(void* const* d_in, const int* in_sizes, int n_in,
                              void* d_out, int out_size, void* d_ws, size_t ws_size,
                              hipStream_t stream) {
  (void)in_sizes; (void)n_in; (void)out_size; (void)ws_size;

  const float* adj = (const float*)d_in[0];
  const float* x   = (const float*)d_in[1];
  const float* Wfm = (const float*)d_in[2];  const float* bfm = (const float*)d_in[3];
  const float* Wc1 = (const float*)d_in[4];  const float* bc1 = (const float*)d_in[5];
  const float* Wp1 = (const float*)d_in[6];  const float* bp1 = (const float*)d_in[7];
  const float* Wc2 = (const float*)d_in[8];  const float* bc2 = (const float*)d_in[9];
  const float* Wp2 = (const float*)d_in[10]; const float* bp2 = (const float*)d_in[11];
  const float* Wc3 = (const float*)d_in[12]; const float* bc3 = (const float*)d_in[13];
  const float* Wh  = (const float*)d_in[14]; const float* bh  = (const float*)d_in[15];
  float* out = (float*)d_out;

  // workspace layout: agg (256KB) | packedXn (128KB) | xT (128KB) | zp (256B)
  char* ws = (char*)d_ws;
  float*    agg      = (float*)ws;
  u32x4*    packedXn = (u32x4*)(ws + (size_t)NN * FF * sizeof(float));
  _Float16* xT       = (_Float16*)(ws + (size_t)NN * FF * sizeof(float)
                                      + (size_t)NN * sizeof(u32x4));
  float*    zpage    = (float*)(ws + (size_t)NN * FF * sizeof(float)
                                   + (size_t)NN * sizeof(u32x4)
                                   + (size_t)NN * FF * sizeof(_Float16));

  qgnn_prep<<<NN / 256, 256, 0, stream>>>(x, packedXn, xT, agg, zpage);
  qgnn_aggregate<<<(ITILES * JSLABS) / 8, 256, 0, stream>>>(
      adj, (const char*)packedXn, xT, (const char*)zpage, agg);
  qgnn_mlp<<<NN / 256, 256, 0, stream>>>(agg, adj, x, Wfm, bfm, Wc1, bc1,
                                         Wp1, bp1, Wc2, bc2, Wp2, bp2,
                                         Wc3, bc3, Wh, bh, out);
}
